// MonkeyJumpRouter_26113401159700
// MI455X (gfx1250) — matrix operand order repaired and checked by reading, not compiled, not run on hardware
//
#include <hip/hip_runtime.h>

typedef __attribute__((ext_vector_type(16))) _Float16     v16h;
typedef __attribute__((ext_vector_type(8)))  float        v8f;
typedef __attribute__((ext_vector_type(4)))  unsigned int u32x4;
typedef __attribute__((ext_vector_type(8)))  int          i32x8;
typedef __attribute__((ext_vector_type(4)))  int          i32x4;

#define H_DIM   2048
#define NEXP    64
#define NTOK    16384          // 4 * 4096
#define MOM     0.9f
#define KC      512            // K-chunk staged in LDS per block
#define KPAD    8              // halves of pad per LDS row (TDM pad: 4 DWORDs)

// --- TDM availability (device pass only; host pass takes the fallback) ------
#if defined(__AMDGCN__) && defined(__gfx1250__) && \
    __has_builtin(__builtin_amdgcn_tensor_load_to_lds) && \
    __has_builtin(__builtin_amdgcn_s_wait_tensorcnt)
  #define HAVE_TDM 1
  #if __has_include(<hip/amd_detail/amd_gfx1250_TDM.h>)
    #define TDM_6ARG 1
  #endif
#endif

__device__ __forceinline__ float wave_reduce_sum(float v) {
#pragma unroll
    for (int off = 16; off > 0; off >>= 1) v += __shfl_xor(v, off, 32);
    return v;
}

// ---------------------------------------------------------------------------
// Kernel 1: normalize centers -> fp16 (WMMA B operand), zero sums/counts.
// ---------------------------------------------------------------------------
__global__ __launch_bounds__(256) void prep_kernel(const float* __restrict__ centers,
                                                   _Float16* __restrict__ cn,
                                                   float* __restrict__ sums,
                                                   float* __restrict__ counts) {
    const int e   = blockIdx.x;
    const int tid = threadIdx.x;
    const float* row = centers + (size_t)e * H_DIM;

    float vals[H_DIM / 256];
    float ss = 0.f;
#pragma unroll
    for (int i = 0; i < H_DIM / 256; ++i) {
        float v = row[tid + 256 * i];
        vals[i] = v;
        ss += v * v;
    }
    ss = wave_reduce_sum(ss);

    __shared__ float red[8];
    __shared__ float s_inv;
    if ((tid & 31) == 0) red[tid >> 5] = ss;
    __syncthreads();
    if (tid == 0) {
        float t = 0.f;
        for (int j = 0; j < 8; ++j) t += red[j];
        s_inv = 1.0f / fmaxf(sqrtf(t), 1e-12f);
    }
    __syncthreads();
    const float inv = s_inv;

#pragma unroll
    for (int i = 0; i < H_DIM / 256; ++i) {
        size_t idx = (size_t)e * H_DIM + tid + 256 * i;
        cn[idx]   = (_Float16)(vals[i] * inv);
        sums[idx] = 0.f;
    }
    if (tid == 0) counts[e] = 0.f;
}

// ---------------------------------------------------------------------------
// Kernel 2: WMMA routing with TDM-staged B operand.
// Per K-chunk of 512: one tensor_load_to_lds pulls [64 x 512] fp16 centers
// into LDS (padded rows, 520 halves stride); 8 waves x 16-token tiles run
// v_wmma_f32_16x16x32_f16 against LDS-resident B fragments.
// ---------------------------------------------------------------------------
__global__ __launch_bounds__(256) void route_kernel(const float* __restrict__ X,
                                                    const _Float16* __restrict__ cn,
                                                    float* __restrict__ outIds,
                                                    float* __restrict__ outW,
                                                    float* __restrict__ sums,
                                                    float* __restrict__ counts) {
    __shared__ _Float16 lds_cn[NEXP][KC + KPAD];   // 64 x 520 halves = 65 KB
    __shared__ float s_logits[8][16][NEXP];        // 32 KB
    __shared__ float s_invn[8][16];
    __shared__ int   s_route[8][16];

    const int wave = threadIdx.x >> 5;
    const int lane = threadIdx.x & 31;
    const int m    = lane & 15;               // token row (A) / expert col (B,C)
    const int hi   = lane >> 4;               // lane half
    const int t0   = (blockIdx.x * 8 + wave) * 16;

    const float* xrow = X + (size_t)(t0 + m) * H_DIM;
    const int koffA = hi ? 16 : 0;            // A-fragment K offset (contiguous k = 16h + s, matches B)
    const int koffB = hi ? 16 : 0;            // B-fragment K offset (contiguous k = 16h + s, matches A)

    v8f acc0 = {0,0,0,0,0,0,0,0};
    v8f acc1 = acc0, acc2 = acc0, acc3 = acc0;
    float sumsq = 0.f;

    for (int kc = 0; kc < H_DIM; kc += KC) {
        __syncthreads();                       // previous chunk fully consumed

#if defined(HAVE_TDM)
        if (threadIdx.x < 32) {                // wave 0 issues the TDM descriptor
            const unsigned long long ga =
                (unsigned long long)(const void*)cn + (unsigned long long)kc * 2ull;
            const unsigned int la = (unsigned int)(unsigned long long)(&lds_cn[0][0]);
            // D# group0: count=1, lds_addr, global_addr[56:0], type=2
            u32x4 g0 = { 1u, la, (unsigned int)ga,
                         (unsigned int)(ga >> 32) | 0x80000000u };
            // D# group1: data_size=2B, pad_enable, pad_interval=256DW, pad_amount=4DW,
            // tensor_dim0=2048, tensor_dim1=64, tile_dim0=512, tile_dim1=64,
            // tensor_dim0_stride=2048
            i32x8 g1 = { 0x07D10000, 0x08000000, 0x00400000, 0x02000000,
                         64, 2048, 0, 0 };
            i32x4 z4 = { 0, 0, 0, 0 };
  #if defined(TDM_6ARG)
            i32x8 z8 = { 0, 0, 0, 0, 0, 0, 0, 0 };
            __builtin_amdgcn_tensor_load_to_lds(g0, g1, z4, z4, z8, 0);
  #else
            __builtin_amdgcn_tensor_load_to_lds(g0, g1, z4, z4, 0);
  #endif
            __builtin_amdgcn_s_wait_tensorcnt(0);
        }
#else
        // Fallback: cooperative staging (16B per transfer, 64 KB total)
        for (int i = threadIdx.x; i < NEXP * (KC / 8); i += 256) {
            const int e  = i / (KC / 8);
            const int c8 = (i % (KC / 8)) * 8;
            *(float4*)&lds_cn[e][c8] = *(const float4*)(cn + (size_t)e * H_DIM + kc + c8);
        }
#endif
        __syncthreads();                       // staged B visible to all waves

        for (int kloc = 0; kloc < KC; kloc += 32) {
            const int k = kc + kloc;
            const float4 a0 = *(const float4*)(xrow + k + koffA);
            const float4 a1 = *(const float4*)(xrow + k + koffA + 4);
            const float4 a2 = *(const float4*)(xrow + k + koffA + 8);
            const float4 a3 = *(const float4*)(xrow + k + koffA + 12);

            __builtin_prefetch(xrow + k + KC, 0, 0);   // next-chunk A prefetch

            sumsq += a0.x*a0.x + a0.y*a0.y + a0.z*a0.z + a0.w*a0.w
                   + a1.x*a1.x + a1.y*a1.y + a1.z*a1.z + a1.w*a1.w
                   + a2.x*a2.x + a2.y*a2.y + a2.z*a2.z + a2.w*a2.w
                   + a3.x*a3.x + a3.y*a3.y + a3.z*a3.z + a3.w*a3.w;

            v16h A;
            A[0]  = (_Float16)a0.x; A[1]  = (_Float16)a0.y;
            A[2]  = (_Float16)a0.z; A[3]  = (_Float16)a0.w;
            A[4]  = (_Float16)a1.x; A[5]  = (_Float16)a1.y;
            A[6]  = (_Float16)a1.z; A[7]  = (_Float16)a1.w;
            A[8]  = (_Float16)a2.x; A[9]  = (_Float16)a2.y;
            A[10] = (_Float16)a2.z; A[11] = (_Float16)a2.w;
            A[12] = (_Float16)a3.x; A[13] = (_Float16)a3.y;
            A[14] = (_Float16)a3.z; A[15] = (_Float16)a3.w;

            const v16h B0 = *(const v16h*)&lds_cn[ 0 + m][kloc + koffB];
            const v16h B1 = *(const v16h*)&lds_cn[16 + m][kloc + koffB];
            const v16h B2 = *(const v16h*)&lds_cn[32 + m][kloc + koffB];
            const v16h B3 = *(const v16h*)&lds_cn[48 + m][kloc + koffB];

            acc0 = __builtin_amdgcn_wmma_f32_16x16x32_f16(false, A, false, B0, (short)0, acc0, false, false);
            acc1 = __builtin_amdgcn_wmma_f32_16x16x32_f16(false, A, false, B1, (short)0, acc1, false, false);
            acc2 = __builtin_amdgcn_wmma_f32_16x16x32_f16(false, A, false, B2, (short)0, acc2, false, false);
            acc3 = __builtin_amdgcn_wmma_f32_16x16x32_f16(false, A, false, B3, (short)0, acc3, false, false);
        }
    }

    // lanes m and m+16 hold disjoint K-halves of token m's sum of squares
    sumsq += __shfl_xor(sumsq, 16, 32);
    const float invn = 1.0f / fmaxf(sqrtf(sumsq), 1e-12f);

    // C/D layout: lanes 0-15 reg r -> M=r; lanes 16-31 reg r -> M=8+r; N = lane&15
#pragma unroll
    for (int r = 0; r < 8; ++r) {
        const int tok = r + 8 * hi;
        s_logits[wave][tok][ 0 + m] = acc0[r];
        s_logits[wave][tok][16 + m] = acc1[r];
        s_logits[wave][tok][32 + m] = acc2[r];
        s_logits[wave][tok][48 + m] = acc3[r];
    }
    if (hi == 0) s_invn[wave][m] = invn;
    __syncthreads();

    if (lane < 16) {
        const int tok = lane;
        const float sc = s_invn[wave][tok];   // TEMPERATURE == 1.0
        float mx = -3.4e38f;
        for (int e = 0; e < NEXP; ++e)
            mx = fmaxf(mx, s_logits[wave][tok][e] * sc);
        float sum = 0.f, b1 = -3.4e38f, b2 = -3.4e38f;
        int i1 = 0, i2 = 0;
        for (int e = 0; e < NEXP; ++e) {
            const float le = s_logits[wave][tok][e] * sc;
            sum += __expf(le - mx);
            if (le > b1)      { b2 = b1; i2 = i1; b1 = le; i1 = e; }
            else if (le > b2) { b2 = le; i2 = e; }
        }
        const float p1 = __expf(b1 - mx) / sum;
        const float p2 = __expf(b2 - mx) / sum;
        const float den = p1 + p2 + 1e-9f;
        const int gt = t0 + tok;
        outIds[gt * 2 + 0] = (float)i1;
        outIds[gt * 2 + 1] = (float)i2;
        outW[gt * 2 + 0] = p1 / den;
        outW[gt * 2 + 1] = p2 / den;
        s_route[wave][tok] = i1;
        atomicAdd(&counts[i1], 1.0f);
    }
    __syncthreads();

    // segment sums of RAW tokens by top-1 expert (X re-read is mostly L2-hot,
    // float atomics resolve at L2)
    for (int tok = 0; tok < 16; ++tok) {
        const int e = s_route[wave][tok];
        const float* xr = X + (size_t)(t0 + tok) * H_DIM;
        float* dst = sums + (size_t)e * H_DIM;
        for (int h = lane; h < H_DIM; h += 32)
            atomicAdd(&dst[h], xr[h]);
    }
}

// ---------------------------------------------------------------------------
// Kernel 3: center update. 64 blocks x 256 threads, one expert per block.
// ---------------------------------------------------------------------------
__global__ __launch_bounds__(256) void finalize_kernel(const float* __restrict__ centers,
                                                       const float* __restrict__ sums,
                                                       const float* __restrict__ counts,
                                                       float* __restrict__ outC) {
    const int e   = blockIdx.x;
    const int tid = threadIdx.x;
    const float cnt = counts[e];

    float cl[H_DIM / 256], vl[H_DIM / 256];
    float ss = 0.f;
#pragma unroll
    for (int i = 0; i < H_DIM / 256; ++i) {
        const size_t idx = (size_t)e * H_DIM + tid + 256 * i;
        const float c = centers[idx];
        const float mean = sums[idx] / (cnt + 1e-6f);
        const float v = MOM * c + (1.0f - MOM) * mean;
        cl[i] = c; vl[i] = v;
        ss += v * v;
    }
    ss = wave_reduce_sum(ss);

    __shared__ float red[8];
    __shared__ float s_inv;
    if ((tid & 31) == 0) red[tid >> 5] = ss;
    __syncthreads();
    if (tid == 0) {
        float t = 0.f;
        for (int j = 0; j < 8; ++j) t += red[j];
        s_inv = 1.0f / fmaxf(sqrtf(t), 1e-12f);
    }
    __syncthreads();
    const float inv = s_inv;
    const bool upd = cnt > 0.f;
#pragma unroll
    for (int i = 0; i < H_DIM / 256; ++i) {
        const size_t idx = (size_t)e * H_DIM + tid + 256 * i;
        outC[idx] = upd ? vl[i] * inv : cl[i];
    }
}

// ---------------------------------------------------------------------------
extern "C" void kernel_launch(void* const* d_in, const int* in_sizes, int n_in,
                              void* d_out, int out_size, void* d_ws, size_t ws_size,
                              hipStream_t stream) {
    const float* X       = (const float*)d_in[0];   // (4, 4096, 2048) f32
    const float* centers = (const float*)d_in[1];   // (64, 2048) f32

    float* out    = (float*)d_out;
    float* outIds = out;                    // 16384*2 ids (as float)
    float* outW   = out + NTOK * 2;         // 16384*2 weights
    float* outC   = out + 2 * NTOK * 2;     // 64*2048 new centers

    char* ws = (char*)d_ws;
    _Float16* cn  = (_Float16*)ws;                                            // 256 KB fp16 centers
    float* sums   = (float*)(ws + (size_t)NEXP * H_DIM * sizeof(_Float16));   // 512 KB
    float* counts = (float*)(ws + (size_t)NEXP * H_DIM * sizeof(_Float16)
                                + (size_t)NEXP * H_DIM * sizeof(float));      // 256 B

    prep_kernel<<<NEXP, 256, 0, stream>>>(centers, cn, sums, counts);
    route_kernel<<<NTOK / 128, 256, 0, stream>>>(X, cn, outIds, outW, sums, counts);
    finalize_kernel<<<NEXP, 256, 0, stream>>>(centers, sums, counts, outC);
}
